// MultiHeadAttention_39410619908561
// MI455X (gfx1250) — matrix operand order repaired and checked by reading, not compiled, not run on hardware
//
#include <hip/hip_runtime.h>

// ---------------------------------------------------------------------------
// Multi-head attention for MI455X (gfx1250), bf16 WMMA pipeline + TDM staging.
//   B=4, S=2048, D=1024, H=16, Hd=64
// Pipeline: [QKV gemm -> bf16 ws] -> [flash attention -> bf16 ctx] -> [out gemm]
// ---------------------------------------------------------------------------

#define D_MODEL   1024
#define NUM_HEADS 16
#define HEAD_DIM  64
#define BATCH     4
#define SEQ       2048
#define MTOT      (BATCH * SEQ)   // 8192

typedef unsigned short ushort_t;
typedef __attribute__((ext_vector_type(16))) __bf16 v16bf;
typedef __attribute__((ext_vector_type(8)))  __bf16 bf16x8;
typedef __attribute__((ext_vector_type(8)))  float  v8f;
typedef __attribute__((ext_vector_type(4)))  unsigned v4u;
typedef __attribute__((ext_vector_type(8)))  int      v8i;
typedef __attribute__((ext_vector_type(4)))  int      v4i;

union Frag {
    v16bf  v;
    bf16x8 h[2];
};

__device__ __forceinline__ ushort_t f2bf(float f) {
    union { float f; unsigned u; } x;
    x.f = f;
    unsigned r = x.u + 0x7FFFu + ((x.u >> 16) & 1u);  // round-to-nearest-even
    return (ushort_t)(r >> 16);
}

__device__ __forceinline__ v8f wmma_bf16(const Frag& a, const Frag& b, v8f c) {
    return __builtin_amdgcn_wmma_f32_16x16x32_bf16(
        /*neg_a=*/false, a.v, /*neg_b=*/false, b.v,
        /*c_mod=*/(short)0, c, /*reuse_a=*/false, /*reuse_b=*/false);
}

// ---------------------------------------------------------------------------
// Tensor Data Mover: 2-D bf16 tile (tile_d0 x tile_d1 elements) from global
// (row stride stride0 elements) into LDS, optionally padding the LDS rows.
// D# layout per CDNA5 ISA.  pad_int/pad_amt are
// the encoded codes (interval DWORDs = 2<<code, amount DWORDs = code+1).
// Must be executed by one wave; completion via s_wait_tensorcnt.
// This toolchain's builtin takes 6 args (g0, g1, g2, g3, g4, cpol).
// ---------------------------------------------------------------------------
__device__ __forceinline__ void tdm_load_2d_bf16(
    void* lds_dst, const void* gsrc,
    unsigned tensor_d0, unsigned tensor_d1,
    unsigned tile_d0, unsigned tile_d1,
    unsigned long long stride0,
    int pad_en, unsigned pad_int, unsigned pad_amt)
{
    const unsigned long long ga = (unsigned long long)(size_t)gsrc;
    const unsigned lds = (unsigned)(size_t)lds_dst;   // generic->LDS offset

    v4u g0;
    g0.x = 1u;                                   // count = 1 valid descriptor
    g0.y = lds;                                  // lds_addr
    g0.z = (unsigned)ga;                         // global_addr[31:0]
    g0.w = (unsigned)(ga >> 32) | (2u << 30);    // global_addr[56:32] | type=2

    unsigned w0 = (1u << 16);                    // data_size = 1 (2 bytes)
    if (pad_en) w0 |= (1u << 20) | (pad_int << 22) | (pad_amt << 25);

    v8i g1;
    g1[0] = (int)w0;                                                  // mask=0, flags
    g1[1] = (int)((tensor_d0 & 0xFFFFu) << 16);                       // dim0 lo
    g1[2] = (int)((tensor_d0 >> 16) | ((tensor_d1 & 0xFFFFu) << 16)); // dim0 hi|dim1 lo
    g1[3] = (int)((tensor_d1 >> 16) | (tile_d0 << 16));               // dim1 hi|tile0
    g1[4] = (int)(tile_d1 & 0xFFFFu);                                 // tile1 (tile2=0)
    g1[5] = (int)(unsigned)(stride0 & 0xFFFFFFFFull);                 // stride0 lo
    g1[6] = (int)(unsigned)(stride0 >> 32);                           // stride0 hi
    g1[7] = 0;
    const v4i gz4 = {0, 0, 0, 0};
    const v8i gz8 = {0, 0, 0, 0, 0, 0, 0, 0};

    __builtin_amdgcn_tensor_load_to_lds(g0, g1, gz4, gz4, gz8, 0);
}

// ============================ GEMM tiling ==================================
#define BM 128
#define BN 128
#define BK 32
#define AST 40   // A LDS row stride (halfs): 32 data + 8 pad, 80B (16B mult)
#define BST 40   // B^T LDS row stride (halfs): row = n, 32 k-values + pad

// ---------------------------------------------------------------------------
// Kernel 1: QKV projection.  out = x @ W + b, written bf16 in [B,H,S,Hd].
// gridDim = (MTOT/BM, D_MODEL/BN, 3);  blockDim = 256 (8 waves, 4x2)
// ---------------------------------------------------------------------------
__global__ __launch_bounds__(256)
void mha_qkv_gemm(const float* __restrict__ X,
                  const float* __restrict__ Wq, const float* __restrict__ bq,
                  const float* __restrict__ Wk, const float* __restrict__ bk,
                  const float* __restrict__ Wv, const float* __restrict__ bv,
                  ushort_t* __restrict__ Q, ushort_t* __restrict__ K,
                  ushort_t* __restrict__ V)
{
    __shared__ ushort_t As[2][BM * AST];
    __shared__ ushort_t Bs[2][BN * BST];

    const float* W; const float* bias; ushort_t* out;
    if (blockIdx.z == 0)      { W = Wq; bias = bq; out = Q; }
    else if (blockIdx.z == 1) { W = Wk; bias = bk; out = K; }
    else                      { W = Wv; bias = bv; out = V; }

    const int tid = threadIdx.x;
    const int m0  = blockIdx.x * BM;
    const int n0  = blockIdx.y * BN;

    const int wid = tid >> 5, lane = tid & 31;
    const int wm  = wid & 3,  wn   = wid >> 2;   // 4 x 2 wave grid
    const int lr  = lane & 15, hi  = lane >> 4;

    const v8f vzero = {0.f, 0.f, 0.f, 0.f, 0.f, 0.f, 0.f, 0.f};
    v8f acc[2][4];
#pragma unroll
    for (int i = 0; i < 2; i++)
#pragma unroll
        for (int j = 0; j < 4; j++) acc[i][j] = vzero;

    auto load_tiles = [&](int t, int buf) {
        const int k0 = t * BK;
        // A tile: 128x32 fp32 -> bf16 LDS (row major, padded)
#pragma unroll
        for (int i = 0; i < 4; i++) {
            int idx = tid + i * 256;
            int r = idx >> 3, c4 = idx & 7;
            const float4 a = *(const float4*)&X[(size_t)(m0 + r) * D_MODEL + k0 + c4 * 4];
            unsigned p0 = (unsigned)f2bf(a.x) | ((unsigned)f2bf(a.y) << 16);
            unsigned p1 = (unsigned)f2bf(a.z) | ((unsigned)f2bf(a.w) << 16);
            *(uint2*)&As[buf][r * AST + c4 * 4] = make_uint2(p0, p1);
        }
        // B tile: W[k0..k0+31][n0..n0+127] fp32 -> transposed bf16 LDS [n][k]
#pragma unroll
        for (int i = 0; i < 4; i++) {
            int idx = tid + i * 256;
            int r = idx >> 5, c4 = idx & 31;
            const float4 w = *(const float4*)&W[(size_t)(k0 + r) * D_MODEL + n0 + c4 * 4];
            Bs[buf][(c4 * 4 + 0) * BST + r] = f2bf(w.x);
            Bs[buf][(c4 * 4 + 1) * BST + r] = f2bf(w.y);
            Bs[buf][(c4 * 4 + 2) * BST + r] = f2bf(w.z);
            Bs[buf][(c4 * 4 + 3) * BST + r] = f2bf(w.w);
        }
    };

    load_tiles(0, 0);
    __syncthreads();

    for (int t = 0; t < D_MODEL / BK; t++) {
        const int cur = t & 1;
        if (t + 1 < D_MODEL / BK) load_tiles(t + 1, cur ^ 1);

        Frag fa[2], fb[4];
#pragma unroll
        for (int mt = 0; mt < 2; mt++) {
            const ushort_t* p = &As[cur][(wm * 32 + mt * 16 + lr) * AST + hi * 16];
            fa[mt].h[0] = *(const bf16x8*)p;
            fa[mt].h[1] = *(const bf16x8*)(p + 8);
        }
#pragma unroll
        for (int nt = 0; nt < 4; nt++) {
            const ushort_t* p = &Bs[cur][(wn * 64 + nt * 16 + lr) * BST + hi * 16];
            fb[nt].h[0] = *(const bf16x8*)p;
            fb[nt].h[1] = *(const bf16x8*)(p + 8);
        }
#pragma unroll
        for (int mt = 0; mt < 2; mt++)
#pragma unroll
            for (int nt = 0; nt < 4; nt++)
                acc[mt][nt] = wmma_bf16(fa[mt], fb[nt], acc[mt][nt]);

        __syncthreads();
    }

    // Write back bf16 to [B,H,S,Hd]
#pragma unroll
    for (int mt = 0; mt < 2; mt++)
#pragma unroll
        for (int nt = 0; nt < 4; nt++) {
            const int n  = n0 + wn * 64 + nt * 16 + lr;
            const int h  = n >> 6, hd = n & 63;
            const float bv_ = bias[n];
#pragma unroll
            for (int i = 0; i < 8; i++) {
                const int m = m0 + wm * 32 + mt * 16 + hi * 8 + i;
                const int b = m >> 11, s = m & 2047;
                out[((size_t)(b * NUM_HEADS + h) * SEQ + s) * HEAD_DIM + hd] =
                    f2bf(acc[mt][nt][i] + bv_);
            }
        }
}

// ---------------------------------------------------------------------------
// Kernel 2: flash attention per (b,h).  q,k,v bf16 [B,H,S,Hd] -> ctx bf16
// [B,S,D].  grid = (SEQ/128, B*H); block = 256 (8 waves, each owns 16 q rows)
// Q and K tiles staged by the Tensor Data Mover (padded 64->72 half rows).
// ---------------------------------------------------------------------------
#define BR  128
#define BC  128
#define QST 72    // 64 + 8 pad (144B rows)  -> TDM pad: 32 DW interval, 4 DW
#define KST 72
#define VST 136   // 128 + 8 pad (272B rows)
#define PST 136

__global__ __launch_bounds__(256)
void mha_attn(const ushort_t* __restrict__ Q, const ushort_t* __restrict__ K,
              const ushort_t* __restrict__ V, ushort_t* __restrict__ CTX)
{
    __shared__ ushort_t Qs[BR * QST];            // 18 KB
    __shared__ ushort_t Ks[BC * KST];            // 18 KB
    __shared__ ushort_t Vt[HEAD_DIM * VST];      // 17 KB (transposed: [hd][ki])
    __shared__ ushort_t Ps[8 * 16 * PST];        // 34 KB (per-wave P rows)

    const int tid = threadIdx.x;
    const int bh  = blockIdx.y;                  // b*16 + h
    const int q0  = blockIdx.x * BR;
    const size_t base = (size_t)bh * SEQ * HEAD_DIM;
    const ushort_t* qp = Q + base;
    const ushort_t* kp = K + base;
    const ushort_t* vp = V + base;

    const int wid = tid >> 5, lane = tid & 31;
    const int lr  = lane & 15, hi  = lane >> 4;

    // Stage Q tile via TDM: tile 64x128 (d x rows), pad rows 64->72 halfs
    if (wid == 0) {
        tdm_load_2d_bf16(Qs, qp + (size_t)q0 * HEAD_DIM,
                         HEAD_DIM, SEQ, HEAD_DIM, BR, HEAD_DIM,
                         /*pad_en=*/1, /*interval 32DW=*/4, /*amount 4DW=*/3);
        __builtin_amdgcn_s_wait_tensorcnt(0);
    }
    __syncthreads();

    // Q A-fragments (wave's 16 rows x K=64 -> 2 fragments), kept in registers
    Frag fq[2];
#pragma unroll
    for (int kk = 0; kk < 2; kk++) {
        const ushort_t* p = &Qs[(wid * 16 + lr) * QST + kk * 32 + hi * 16];
        fq[kk].h[0] = *(const bf16x8*)p;
        fq[kk].h[1] = *(const bf16x8*)(p + 8);
    }

    const v8f vzero = {0.f, 0.f, 0.f, 0.f, 0.f, 0.f, 0.f, 0.f};
    v8f oacc[4];
    float m_run[8], l_run[8];
#pragma unroll
    for (int i = 0; i < 4; i++) oacc[i] = vzero;
#pragma unroll
    for (int i = 0; i < 8; i++) { m_run[i] = -3.0e38f; l_run[i] = 0.f; }

    const float SCL2E = 0.125f * 1.44269504088896f;  // 1/sqrt(64) * log2(e)

    for (int kt = 0; kt < SEQ / BC; kt++) {
        __syncthreads();  // previous iter's Vt/Ks reads complete before refill

        // K tile via TDM: [ki 0..127][d 0..63] row-major == WMMA B operand [n][k]
        if (wid == 0)
            tdm_load_2d_bf16(Ks, kp + (size_t)(kt * BC) * HEAD_DIM,
                             HEAD_DIM, SEQ, HEAD_DIM, BC, HEAD_DIM,
                             1, 4, 3);
        // V tile transposed into Vt[hd][ki] (manual: TDM cannot transpose)
#pragma unroll
        for (int i = 0; i < 4; i++) {
            int idx = tid + i * 256;
            int r = idx >> 3, c = idx & 7;
            uint4 d = *(const uint4*)&vp[(size_t)(kt * BC + r) * HEAD_DIM + c * 8];
            const ushort_t* hsrc = (const ushort_t*)&d;
#pragma unroll
            for (int j = 0; j < 8; j++) Vt[(c * 8 + j) * VST + r] = hsrc[j];
        }
        if (kt + 1 < SEQ / BC)
            __builtin_prefetch(&vp[(size_t)((kt + 1) * BC + (tid & 127)) * HEAD_DIM], 0, 1);
        if (wid == 0) __builtin_amdgcn_s_wait_tensorcnt(0);
        __syncthreads();

        // ---- scores: wave computes S[16 q rows][128 ki] ----
        v8f sacc[8];
#pragma unroll
        for (int nt = 0; nt < 8; nt++) sacc[nt] = vzero;
#pragma unroll
        for (int nt = 0; nt < 8; nt++) {
            const ushort_t* p = &Ks[(nt * 16 + lr) * KST + hi * 16];
            Frag fk0, fk1;
            fk0.h[0] = *(const bf16x8*)p;         fk0.h[1] = *(const bf16x8*)(p + 8);
            fk1.h[0] = *(const bf16x8*)(p + 32);  fk1.h[1] = *(const bf16x8*)(p + 40);
            sacc[nt] = wmma_bf16(fq[0], fk0, sacc[nt]);
            sacc[nt] = wmma_bf16(fq[1], fk1, sacc[nt]);
        }

        // ---- online softmax (row r = hi*8 + i lives in one 16-lane group) ----
        float alpha[8];
#pragma unroll
        for (int i = 0; i < 8; i++) {
            float mx = sacc[0][i];
#pragma unroll
            for (int nt = 1; nt < 8; nt++) mx = fmaxf(mx, sacc[nt][i]);
            mx = fmaxf(mx, __shfl_xor(mx, 1, 16));
            mx = fmaxf(mx, __shfl_xor(mx, 2, 16));
            mx = fmaxf(mx, __shfl_xor(mx, 4, 16));
            mx = fmaxf(mx, __shfl_xor(mx, 8, 16));
            const float mnew = fmaxf(m_run[i], mx);
            const float a    = exp2f((m_run[i] - mnew) * SCL2E);
            alpha[i] = a;
            float ls = 0.f;
#pragma unroll
            for (int nt = 0; nt < 8; nt++) {
                const float pv = exp2f((sacc[nt][i] - mnew) * SCL2E);
                sacc[nt][i] = pv;
                ls += pv;
            }
            ls += __shfl_xor(ls, 1, 16);
            ls += __shfl_xor(ls, 2, 16);
            ls += __shfl_xor(ls, 4, 16);
            ls += __shfl_xor(ls, 8, 16);
            l_run[i] = l_run[i] * a + ls;
            m_run[i] = mnew;
#pragma unroll
            for (int nt = 0; nt < 4; nt++) oacc[nt][i] *= a;
        }

        // ---- reshape P (C-layout -> A-fragments) through this wave's LDS ----
        const int pbase = wid * 16 * PST;
#pragma unroll
        for (int nt = 0; nt < 8; nt++)
#pragma unroll
            for (int i = 0; i < 8; i++)
                Ps[pbase + (hi * 8 + i) * PST + nt * 16 + lr] = f2bf(sacc[nt][i]);
        // same-wave LDS ops are in-order; no barrier needed (private region)

        // ---- O += P @ V ----
#pragma unroll
        for (int kk = 0; kk < 4; kk++) {
            Frag fp;
            const ushort_t* pp = &Ps[pbase + lr * PST + kk * 32 + hi * 16];
            fp.h[0] = *(const bf16x8*)pp;
            fp.h[1] = *(const bf16x8*)(pp + 8);
#pragma unroll
            for (int nt = 0; nt < 4; nt++) {
                Frag fv;
                const ushort_t* pv = &Vt[(nt * 16 + lr) * VST + kk * 32 + hi * 16];
                fv.h[0] = *(const bf16x8*)pv;
                fv.h[1] = *(const bf16x8*)(pv + 8);
                oacc[nt] = wmma_bf16(fp, fv, oacc[nt]);
            }
        }
    }

    // ---- normalize and write ctx bf16 [B,S,D] ----
    const int b = bh >> 4, h = bh & 15;
#pragma unroll
    for (int nt = 0; nt < 4; nt++) {
        const int hd = nt * 16 + lr;
#pragma unroll
        for (int i = 0; i < 8; i++) {
            const int srow = q0 + wid * 16 + hi * 8 + i;
            const float r  = oacc[nt][i] / l_run[i];
            CTX[((size_t)(b * SEQ + srow)) * D_MODEL + h * HEAD_DIM + hd] = f2bf(r);
        }
    }
}

// ---------------------------------------------------------------------------
// Kernel 3: output projection.  out = ctx(bf16) @ Wo + bo  (fp32 out)
// grid = (MTOT/BM, D_MODEL/BN); block = 256.  A tiles staged by TDM
// (double-buffered: DMA of tile t+1 overlaps WMMA on tile t).
// ---------------------------------------------------------------------------
__global__ __launch_bounds__(256)
void mha_out_gemm(const ushort_t* __restrict__ A, const float* __restrict__ W,
                  const float* __restrict__ bias, float* __restrict__ OUT)
{
    __shared__ ushort_t As[2][BM * AST];
    __shared__ ushort_t Bs[2][BN * BST];

    const int tid = threadIdx.x;
    const int m0  = blockIdx.x * BM;
    const int n0  = blockIdx.y * BN;

    const int wid = tid >> 5, lane = tid & 31;
    const int wm  = wid & 3,  wn   = wid >> 2;
    const int lr  = lane & 15, hi  = lane >> 4;

    const v8f vzero = {0.f, 0.f, 0.f, 0.f, 0.f, 0.f, 0.f, 0.f};
    v8f acc[2][4];
#pragma unroll
    for (int i = 0; i < 2; i++)
#pragma unroll
        for (int j = 0; j < 4; j++) acc[i][j] = vzero;

    // TDM: A tile [128 rows x 32 halfs], row stride 1024, pad 32->40 halfs
    auto load_a = [&](int t, int buf) {
        tdm_load_2d_bf16(&As[buf][0], &A[(size_t)m0 * D_MODEL + t * BK],
                         D_MODEL, MTOT, BK, BM, D_MODEL,
                         /*pad_en=*/1, /*interval 16DW=*/3, /*amount 4DW=*/3);
    };
    auto load_b = [&](int t, int buf) {
        const int k0 = t * BK;
#pragma unroll
        for (int i = 0; i < 4; i++) {
            int idx = tid + i * 256;
            int r = idx >> 5, c4 = idx & 31;
            const float4 w = *(const float4*)&W[(size_t)(k0 + r) * D_MODEL + n0 + c4 * 4];
            Bs[buf][(c4 * 4 + 0) * BST + r] = f2bf(w.x);
            Bs[buf][(c4 * 4 + 1) * BST + r] = f2bf(w.y);
            Bs[buf][(c4 * 4 + 2) * BST + r] = f2bf(w.z);
            Bs[buf][(c4 * 4 + 3) * BST + r] = f2bf(w.w);
        }
    };

    if (wid == 0) load_a(0, 0);
    load_b(0, 0);
    if (wid == 0) __builtin_amdgcn_s_wait_tensorcnt(0);
    __syncthreads();

    for (int t = 0; t < D_MODEL / BK; t++) {
        const int cur = t & 1;
        if (t + 1 < D_MODEL / BK) {
            if (wid == 0) load_a(t + 1, cur ^ 1);
            load_b(t + 1, cur ^ 1);
        }

        Frag fa[2], fb[4];
#pragma unroll
        for (int mt = 0; mt < 2; mt++) {
            const ushort_t* p = &As[cur][(wm * 32 + mt * 16 + lr) * AST + hi * 16];
            fa[mt].h[0] = *(const bf16x8*)p;
            fa[mt].h[1] = *(const bf16x8*)(p + 8);
        }
#pragma unroll
        for (int nt = 0; nt < 4; nt++) {
            const ushort_t* p = &Bs[cur][(wn * 64 + nt * 16 + lr) * BST + hi * 16];
            fb[nt].h[0] = *(const bf16x8*)p;
            fb[nt].h[1] = *(const bf16x8*)(p + 8);
        }
#pragma unroll
        for (int mt = 0; mt < 2; mt++)
#pragma unroll
            for (int nt = 0; nt < 4; nt++)
                acc[mt][nt] = wmma_bf16(fa[mt], fb[nt], acc[mt][nt]);

        if (wid == 0) __builtin_amdgcn_s_wait_tensorcnt(0);
        __syncthreads();
    }

#pragma unroll
    for (int mt = 0; mt < 2; mt++)
#pragma unroll
        for (int nt = 0; nt < 4; nt++) {
            const int n = n0 + wn * 64 + nt * 16 + lr;
            const float bo_ = bias[n];
#pragma unroll
            for (int i = 0; i < 8; i++) {
                const int m = m0 + wm * 32 + mt * 16 + hi * 8 + i;
                OUT[(size_t)m * D_MODEL + n] = acc[mt][nt][i] + bo_;
            }
        }
}

// ---------------------------------------------------------------------------
extern "C" void kernel_launch(void* const* d_in, const int* in_sizes, int n_in,
                              void* d_out, int out_size, void* d_ws, size_t ws_size,
                              hipStream_t stream)
{
    const float* x  = (const float*)d_in[0];
    const float* Wq = (const float*)d_in[1];
    const float* bq = (const float*)d_in[2];
    const float* Wk = (const float*)d_in[3];
    const float* bk = (const float*)d_in[4];
    const float* Wv = (const float*)d_in[5];
    const float* bv = (const float*)d_in[6];
    const float* Wo = (const float*)d_in[7];
    const float* bo = (const float*)d_in[8];
    float* out = (float*)d_out;

    const size_t NELEM = (size_t)MTOT * D_MODEL;  // 8.39M elements
    ushort_t* qws = (ushort_t*)d_ws;
    ushort_t* kws = qws + NELEM;
    ushort_t* vws = kws + NELEM;
    ushort_t* cws = vws + NELEM;

    dim3 g1(MTOT / BM, D_MODEL / BN, 3);
    mha_qkv_gemm<<<g1, 256, 0, stream>>>(x, Wq, bq, Wk, bk, Wv, bv, qws, kws, vws);

    dim3 g2(SEQ / BR, BATCH * NUM_HEADS);
    mha_attn<<<g2, 256, 0, stream>>>(qws, kws, vws, cws);

    dim3 g3(MTOT / BM, D_MODEL / BN);
    mha_out_gemm<<<g3, 256, 0, stream>>>(cws, Wo, bo, out);
}
